// PyGRandLANet_84086869721222
// MI455X (gfx1250) — matrix operand order repaired and checked by reading, not compiled, not run on hardware
//
#include <hip/hip_runtime.h>

// ============================================================================
// RandLA-Net forward for MI455X (gfx1250, wave32).
// All Linear layers run on the WMMA pipe: v_wmma_f32_16x16x32_f16, f16
// operands staged via LDS (B transposed so fragments are contiguous b128
// loads), fp32 accumulation.  BatchNorm folded to per-channel scale/shift.
// knn_interpolate(k=1) == NN gather.  Deterministic: no atomics.
// ============================================================================

typedef __attribute__((ext_vector_type(16))) _Float16 v16h;
typedef __attribute__((ext_vector_type(8)))  _Float16 v8h;
typedef __attribute__((ext_vector_type(4)))  _Float16 v4h;
typedef __attribute__((ext_vector_type(8)))  float    v8f;

#define EPS_BN 1e-6f

// ---------------------------------------------------------------------------
// WMMA GEMM:  C[M,N] = A[M,K] * W[K,N] (+bias) (+optional leaky-relu)
// A,W,C fp32 row-major.  One 16x16 tile per wave, 4 waves/block (64 rows),
// K-tile = 64 (two v_wmma per stage).  Clamped loads (no divergent guards).
// ---------------------------------------------------------------------------
__global__ __launch_bounds__(128)
void k_gemm(const float* __restrict__ A, const float* __restrict__ W,
            const float* __restrict__ bias, float* __restrict__ C,
            int M, int N, int K, int act, float slope) {
  __shared__ alignas(16) _Float16 Al[4][16][64];   // per-wave A tile [m][k]
  __shared__ alignas(16) _Float16 Bt[16][64];      // shared B tile, transposed [n][k]
  const int tid  = threadIdx.x;
  const int lane = tid & 31;
  const int w    = tid >> 5;
  const int col0 = blockIdx.x * 16;
  const int row0 = blockIdx.y * 64 + w * 16;
  const bool a4  = ((K & 3) == 0);

  v8f acc = {};
  for (int k0 = 0; k0 < K; k0 += 64) {
    // ---- stage A tile (16x64): lane -> row lane&15, k-half (lane>>4)*32
    {
      const int m  = lane & 15;
      const int kb = (lane >> 4) * 32;
      const int gr = row0 + m;
      const bool rok = gr < M;
      const long rbase = (long)(rok ? gr : (M - 1)) * K;
      if (a4 && rok && (k0 + 64 <= K)) {           // fast: aligned float4 loads
        const float4* src = (const float4*)(A + rbase + k0 + kb);
#pragma unroll
        for (int q = 0; q < 8; ++q) {
          float4 f = src[q];
          v4h h = {(_Float16)f.x, (_Float16)f.y, (_Float16)f.z, (_Float16)f.w};
          *(v4h*)&Al[w][m][kb + q * 4] = h;
        }
      } else {                                      // tail: clamped + cndmask
#pragma unroll
        for (int q = 0; q < 32; ++q) {
          int kk = k0 + kb + q;
          float v = A[rbase + ((kk < K) ? kk : (K - 1))];
          v = (rok && kk < K) ? v : 0.f;
          Al[w][m][kb + q] = (_Float16)v;
        }
      }
    }
    // ---- stage B tile transposed (64k x 16n): 128 thr * 8 elems
    {
      const int k  = tid >> 1;            // 0..63
      const int n8 = (tid & 1) * 8;       // 0 or 8
      const int kk = k0 + k;
      const long rb = (long)((kk < K) ? kk : (K - 1)) * N;
#pragma unroll
      for (int q = 0; q < 8; ++q) {
        int gc = col0 + n8 + q;
        float v = W[rb + ((gc < N) ? gc : (N - 1))];
        v = (kk < K && gc < N) ? v : 0.f;
        Bt[n8 + q][k] = (_Float16)v;
      }
    }
    __syncthreads();
    // ---- two 16x16x32 WMMAs; fragments are contiguous b128 LDS loads
    const int mrow = lane & 15;
    const int khiA = (lane & 16) >> 1;    // +8  for lanes 16..31
    const int khiB = lane & 16;           // +16 for lanes 16..31
#pragma unroll
    for (int half = 0; half < 2; ++half) {
      const int kb = half * 32;
      v8h a0 = *(const v8h*)&Al[w][mrow][kb + khiA];
      v8h a1 = *(const v8h*)&Al[w][mrow][kb + 16 + khiA];
      v8h b0 = *(const v8h*)&Bt[mrow][kb + khiA];
      v8h b1 = *(const v8h*)&Bt[mrow][kb + 16 + khiA];
      v16h af = __builtin_shufflevector(a0, a1, 0, 1, 2, 3, 4, 5, 6, 7,
                                        8, 9, 10, 11, 12, 13, 14, 15);
      v16h bf = __builtin_shufflevector(b0, b1, 0, 1, 2, 3, 4, 5, 6, 7,
                                        8, 9, 10, 11, 12, 13, 14, 15);
      acc = __builtin_amdgcn_wmma_f32_16x16x32_f16(
          false, af, false, bf, (short)0, acc, false, false);
    }
    __syncthreads();
  }
  // ---- store: C/D layout — row = r + 8*(lane>=16), col = lane&15
  const int n = lane & 15;
  const int gcol = col0 + n;
  if (gcol < N) {
    float bv = bias ? bias[gcol] : 0.f;
#pragma unroll
    for (int r = 0; r < 8; ++r) {
      int grow = row0 + r + ((lane & 16) >> 1);
      if (grow < M) {
        float v = acc[r] + bv;
        if (act) v = v >= 0.f ? v : slope * v;
        C[(long)grow * N + gcol] = v;
      }
    }
  }
}

// ---------------------------------------------------------------------------
// BatchNorm batch-stats -> fused scale/shift per channel (one block/channel)
// ---------------------------------------------------------------------------
__global__ void k_colstats(const float* __restrict__ X, int R, int C,
                           const float* __restrict__ g, const float* __restrict__ bt,
                           float eps, float* __restrict__ scale,
                           float* __restrict__ shift) {
  const int c = blockIdx.x;
  __shared__ float s1[256], s2[256];
  float a = 0.f, b = 0.f;
  for (int r = threadIdx.x; r < R; r += 256) {
    float v = X[(long)r * C + c];
    a += v; b += v * v;
  }
  s1[threadIdx.x] = a; s2[threadIdx.x] = b;
  __syncthreads();
  for (int o = 128; o; o >>= 1) {
    if (threadIdx.x < o) { s1[threadIdx.x] += s1[threadIdx.x + o];
                           s2[threadIdx.x] += s2[threadIdx.x + o]; }
    __syncthreads();
  }
  if (threadIdx.x == 0) {
    float m = s1[0] / (float)R;
    float v = fmaxf(s2[0] / (float)R - m * m, 0.f);
    float sc = g[c] * rsqrtf(v + eps);
    scale[c] = sc;
    shift[c] = bt[c] - m * sc;
  }
}

__global__ void k_bn_act(float* __restrict__ X, long tot, int C,
                         const float* __restrict__ scale,
                         const float* __restrict__ shift, int act, float slope) {
  long i = (long)blockIdx.x * 256 + threadIdx.x;
  if (i >= tot) return;
  int c = (int)(i % C);
  float v = X[i] * scale[c] + shift[c];
  if (act) v = v >= 0.f ? v : slope * v;
  X[i] = v;
}

// ---------------------------------------------------------------------------
// Brute-force KNN: queries pos[4*m] (DEC=4), K=16, ties -> lower index.
// Branchless unrolled register insertion (cndmask chains, no scratch).
// ---------------------------------------------------------------------------
__global__ void k_knn(const float* __restrict__ pos, int n, int M,
                      int* __restrict__ src) {
  int m = blockIdx.x * 128 + threadIdx.x;
  if (m >= M) return;
  const float qx = pos[3 * (m * 4) + 0];
  const float qy = pos[3 * (m * 4) + 1];
  const float qz = pos[3 * (m * 4) + 2];
  float bd[16]; int bi[16];
#pragma unroll
  for (int t = 0; t < 16; ++t) { bd[t] = 3.0e38f; bi[t] = 0; }
  for (int j = 0; j < n; ++j) {
    float dx = pos[3 * j + 0] - qx;
    float dy = pos[3 * j + 1] - qy;
    float dz = pos[3 * j + 2] - qz;
    float d2 = dx * dx + dy * dy + dz * dz;
    if (d2 < bd[15]) {
#pragma unroll
      for (int t = 15; t >= 1; --t) {
        bool up  = bd[t - 1] > d2;              // insertion point above -> shift
        bool ins = !up && (d2 < bd[t]);         // insert exactly here
        float nv = up ? bd[t - 1] : (ins ? d2 : bd[t]);
        int   ni = up ? bi[t - 1] : (ins ? j  : bi[t]);
        bd[t] = nv; bi[t] = ni;
      }
      if (d2 < bd[0]) { bd[0] = d2; bi[0] = j; }
    }
  }
  for (int t = 0; t < 16; ++t) src[m * 16 + t] = bi[t];
}

// relative point encoding + gather x_j into lf[:, 0:dx]
// (faithful quirk: pos_i = pos[dst] = pos[e/16], NOT decimated positions)
__global__ void k_rel(const float* __restrict__ pos, const float* __restrict__ x,
                      int dx, const int* __restrict__ src, int E,
                      float* __restrict__ rel, float* __restrict__ lf, int dlf) {
  int e = blockIdx.x * 256 + threadIdx.x;
  if (e >= E) return;
  int i = e >> 4;
  int j = src[e];
  float pix = pos[3 * i], piy = pos[3 * i + 1], piz = pos[3 * i + 2];
  float pjx = pos[3 * j], pjy = pos[3 * j + 1], pjz = pos[3 * j + 2];
  float dxv = pjx - pix, dyv = pjy - piy, dzv = pjz - piz;
  float eu = fabsf(dxv) + fabsf(dyv) + fabsf(dzv);   // sum(sqrt(d*d)) == L1
  float* r = rel + (long)e * 10;
  r[0] = pix; r[1] = piy; r[2] = piz;
  r[3] = pjx; r[4] = pjy; r[5] = pjz;
  r[6] = dxv; r[7] = dyv; r[8] = dzv; r[9] = eu;
  const float* xr = x + (long)j * dx;
  float* o = lf + (long)e * dlf;
  for (int c = 0; c < dx; ++c) o[c] = xr[c];
}

__global__ void k_copycols(const float* __restrict__ S, int Cs,
                           float* __restrict__ D, int Cd, int off, long tot) {
  long i = (long)blockIdx.x * 256 + threadIdx.x;
  if (i >= tot) return;
  long r = i / Cs; int c = (int)(i % Cs);
  D[r * Cd + off + c] = S[i];
}

// row-wise softmax, one wave per row
__global__ void k_rowsoftmax(float* __restrict__ X, int R, int C) {
  int row = blockIdx.x * 8 + (threadIdx.x >> 5);
  int lane = threadIdx.x & 31;
  if (row >= R) return;
  float* x = X + (long)row * C;
  float mx = -3.0e38f;
  for (int c = lane; c < C; c += 32) mx = fmaxf(mx, x[c]);
  for (int o = 16; o; o >>= 1) mx = fmaxf(mx, __shfl_xor(mx, o, 32));
  float s = 0.f;
  for (int c = lane; c < C; c += 32) { float e = __expf(x[c] - mx); x[c] = e; s += e; }
  for (int o = 16; o; o >>= 1) s += __shfl_xor(s, o, 32);
  float inv = 1.f / s;
  for (int c = lane; c < C; c += 32) x[c] *= inv;
}

// segment_sum: dst = e/16, fixed segment size 16 -> exact, no atomics
__global__ void k_segsum(const float* __restrict__ att, const float* __restrict__ lf,
                         int d, int M, float* __restrict__ agg) {
  int i = blockIdx.x * 256 + threadIdx.x;
  if (i >= M * d) return;
  int m = i / d, c = i % d;
  long base = (long)m * 16 * d + c;
  float s = 0.f;
#pragma unroll
  for (int t = 0; t < 16; ++t) s += att[base + (long)t * d] * lf[base + (long)t * d];
  agg[(long)m * d + c] = s;
}

__global__ void k_add_lrelu(const float* __restrict__ a, const float* __restrict__ b,
                            float* __restrict__ o, long tot, float slope) {
  long i = (long)blockIdx.x * 256 + threadIdx.x;
  if (i >= tot) return;
  float v = a[i] + b[i];
  o[i] = v >= 0.f ? v : slope * v;
}

__global__ void k_cat2(const float* __restrict__ a, int Ca,
                       const float* __restrict__ b, int Cb,
                       float* __restrict__ o, long tot) {
  long i = (long)blockIdx.x * 256 + threadIdx.x;
  if (i >= tot) return;
  int Ct = Ca + Cb;
  long r = i / Ct; int c = (int)(i % Ct);
  o[i] = (c < Ca) ? a[r * Ca + c] : b[r * Cb + (c - Ca)];
}

// Y[m] = X[4m]  (idx = arange(0,n,DEC))
__global__ void k_decimate(const float* __restrict__ X, int C,
                           float* __restrict__ Y, long tot) {
  long i = (long)blockIdx.x * 256 + threadIdx.x;
  if (i >= tot) return;
  long r = i / C; int c = (int)(i % C);
  Y[i] = X[(r * 4) * C + c];
}

// knn_interpolate(k=1): weights cancel -> pure nearest-neighbor index
__global__ void k_nn1(const float* __restrict__ posC, int Nc,
                      const float* __restrict__ posS, int Ns, int* __restrict__ nn) {
  int s = blockIdx.x * 128 + threadIdx.x;
  if (s >= Ns) return;
  float qx = posS[3 * s], qy = posS[3 * s + 1], qz = posS[3 * s + 2];
  float best = 3.0e38f; int bi = 0;
  for (int j = 0; j < Nc; ++j) {
    float dx = posC[3 * j] - qx, dy = posC[3 * j + 1] - qy, dz = posC[3 * j + 2] - qz;
    float d2 = dx * dx + dy * dy + dz * dz;
    if (d2 < best) { best = d2; bi = j; }
  }
  nn[s] = bi;
}

// out[s] = concat(xc[nn[s]], xs[s])
__global__ void k_fpcat(const float* __restrict__ xc, int Cc,
                        const int* __restrict__ nn,
                        const float* __restrict__ xs, int Cs,
                        float* __restrict__ o, long tot) {
  long i = (long)blockIdx.x * 256 + threadIdx.x;
  if (i >= tot) return;
  int Ct = Cc + Cs;
  long r = i / Ct; int c = (int)(i % Ct);
  o[i] = (c < Cc) ? xc[(long)nn[r] * Cc + c] : xs[r * Cs + (c - Cc)];
}

// ===========================================================================
// Host-side orchestration
// ===========================================================================
struct Lin  { const float *W, *b, *bt, *g; };
struct LfaP { Lin att, enc, post; };
struct BlkP { LfaP l1, l2; Lin m1, m2, sc; };

struct Bump {
  char* base; size_t off;
  float* f(size_t n) {
    float* p = (float*)(base + off);
    off += ((n * 4 + 255) & ~(size_t)255);
    return p;
  }
  int* i(size_t n) { return (int*)f(n); }
};

static inline unsigned gb(long tot) { return (unsigned)((tot + 255) / 256); }

static void linear(const float* A, const float* W, const float* b, float* C,
                   int M, int N, int K, int act, float slope, hipStream_t s) {
  dim3 g((N + 15) / 16, (M + 63) / 64);
  k_gemm<<<g, 128, 0, s>>>(A, W, b, C, M, N, K, act, slope);
}

static void mlp_bn(const float* A, const Lin& L, float* C, int M, int N, int K,
                   int act, float slope, float eps, float* stat, hipStream_t s) {
  linear(A, L.W, L.b, C, M, N, K, 0, 0.f, s);
  k_colstats<<<N, 256, 0, s>>>(C, M, N, L.g, L.bt, eps, stat, stat + N);
  long tot = (long)M * N;
  k_bn_act<<<gb(tot), 256, 0, s>>>(C, tot, N, stat, stat + N, act, slope);
}

static void lfa(const LfaP& p, const int* src, const float* x, int dx,
                const float* pos, int n, int M, float* out, Bump& ws,
                float* stat, hipStream_t s) {
  const int d = 2 * dx, E = M * 16;
  float* rel = ws.f((size_t)E * 10);
  float* lf  = ws.f((size_t)E * d);
  float* enc = ws.f((size_t)E * dx);
  float* att = ws.f((size_t)E * d);
  float* agg = ws.f((size_t)n * d);
  k_rel<<<gb(E), 256, 0, s>>>(pos, x, dx, src, E, rel, lf, d);
  mlp_bn(rel, p.enc, enc, E, dx, 10, 1, 0.2f, EPS_BN, stat, s);
  k_copycols<<<gb((long)E * dx), 256, 0, s>>>(enc, dx, lf, d, dx, (long)E * dx);
  linear(lf, p.att.W, nullptr, att, E, d, d, 0, 0.f, s);
  k_rowsoftmax<<<(E + 7) / 8, 256, 0, s>>>(att, E, d);
  hipMemsetAsync(agg, 0, (size_t)n * d * 4, s);
  k_segsum<<<gb((long)M * d), 256, 0, s>>>(att, lf, d, M, agg);
  mlp_bn(agg, p.post, out, n, d, d, 1, 0.2f, EPS_BN, stat, s);
}

static void block(const BlkP& p, const float* x, const float* pos, int n,
                  int din, int dout, float* out, Bump& ws, float* stat,
                  hipStream_t s) {
  const int M = n / 4;
  int*   src = ws.i((size_t)M * 16);
  float* sc  = ws.f((size_t)n * 2 * dout);
  float* h   = ws.f((size_t)n * (dout / 4));
  float* l1  = ws.f((size_t)n * (dout / 2));
  float* l2  = ws.f((size_t)n * dout);
  float* m2  = ws.f((size_t)n * 2 * dout);
  k_knn<<<(M + 127) / 128, 128, 0, s>>>(pos, n, M, src);
  mlp_bn(x, p.sc, sc, n, 2 * dout, din, 0, 0.f, EPS_BN, stat, s);        // shortcut
  linear(x, p.m1.W, p.m1.b, h, n, dout / 4, din, 1, 0.2f, s);            // mlp1
  size_t mk = ws.off;
  lfa(p.l1, src, h, dout / 4, pos, n, M, l1, ws, stat, s);
  ws.off = mk;
  lfa(p.l2, src, l1, dout / 2, pos, n, M, l2, ws, stat, s);
  ws.off = mk;
  mlp_bn(l2, p.m2, m2, n, 2 * dout, dout, 0, 0.f, EPS_BN, stat, s);      // mlp2
  long tot = (long)n * 2 * dout;
  k_add_lrelu<<<gb(tot), 256, 0, s>>>(m2, sc, out, tot, 0.01f);          // residual
}

static void fp_up(const Lin& L, const float* xc, int Cc, const float* posC, int Nc,
                  const float* posS, int Ns, const float* xs, int Cs,
                  float* out, int Cout, Bump& ws, float* stat, hipStream_t s) {
  int*   nn  = ws.i((size_t)Ns);
  float* cat = ws.f((size_t)Ns * (Cc + Cs));
  k_nn1<<<(Ns + 127) / 128, 128, 0, s>>>(posC, Nc, posS, Ns, nn);
  long tot = (long)Ns * (Cc + Cs);
  k_fpcat<<<gb(tot), 256, 0, s>>>(xc, Cc, nn, xs, Cs, cat, tot);
  mlp_bn(cat, L, out, Ns, Cout, Cc + Cs, 1, 0.2f, EPS_BN, stat, s);
}

extern "C" void kernel_launch(void* const* d_in, const int* in_sizes, int n_in,
                              void* d_out, int out_size, void* d_ws, size_t ws_size,
                              hipStream_t stream) {
  (void)in_sizes; (void)n_in; (void)out_size; (void)ws_size;
  // ---- decode params in canonical JAX pytree (sorted-key) order ----
  int t = 0;
  auto nx = [&]() { return (const float*)d_in[t++]; };
  auto rd_att = [&]() { Lin l{nullptr, nullptr, nullptr, nullptr}; l.W = nx(); return l; };
  auto rd_lin = [&]() { Lin l{nullptr, nullptr, nullptr, nullptr}; l.W = nx(); l.b = nx(); return l; };
  auto rd_bn  = [&]() { Lin l{}; l.W = nx(); l.b = nx(); l.bt = nx(); l.g = nx(); return l; };
  auto rd_lfa = [&]() { LfaP p; p.att = rd_att(); p.enc = rd_bn(); p.post = rd_bn(); return p; };
  auto rd_blk = [&]() { BlkP b; b.l1 = rd_lfa(); b.l2 = rd_lfa();
                        b.m1 = rd_lin(); b.m2 = rd_bn(); b.sc = rd_bn(); return b; };
  BlkP B1 = rd_blk(), B2 = rd_blk(), B3 = rd_blk(), B4 = rd_blk();
  Lin FC0 = rd_bn();
  Lin FCE = rd_lin();
  Lin FP1 = rd_bn(), FP2 = rd_bn(), FP3 = rd_bn(), FP4 = rd_bn();
  Lin H1 = rd_bn(), H2 = rd_bn(), MID = rd_bn();
  const float* pos = nx();
  const float* x   = nx();
  hipStream_t s = stream;

  // ---- workspace (deterministic bump allocator, reset every call) ----
  Bump ws{(char*)d_ws, 0};
  float* stat = ws.f(1024);                       // fused BN scale|shift (<=512ch)
  float* h0   = ws.f((size_t)8192 * 16);
  float* x1f  = ws.f((size_t)8192 * 32);
  float* pos1 = ws.f((size_t)2048 * 3);  float* x1 = ws.f((size_t)2048 * 32);
  float* pos2 = ws.f((size_t)512 * 3);   float* x2 = ws.f((size_t)512 * 128);
  float* pos3 = ws.f((size_t)128 * 3);   float* x3 = ws.f((size_t)128 * 256);
  float* pos4 = ws.f((size_t)32 * 3);    float* x4 = ws.f((size_t)32 * 512);
  float* mm   = ws.f((size_t)32 * 512);
  float* f4   = ws.f((size_t)128 * 256);
  float* f3   = ws.f((size_t)512 * 128);
  float* f2   = ws.f((size_t)8192 * 32);
  float* f1   = ws.f((size_t)8192 * 8);
  float* g1   = ws.f((size_t)8192 * 64);
  float* g2   = ws.f((size_t)8192 * 32);
  const size_t mark = ws.off;

  auto dec = [&](const float* X, int C, float* Y, int M) {
    long tot = (long)M * C;
    k_decimate<<<gb(tot), 256, 0, s>>>(X, C, Y, tot);
  };

  // fc0: bnorm(concat([pos, x]) @ W + b)   (no activation)
  {
    float* in0 = ws.f((size_t)8192 * 9);
    k_cat2<<<gb((long)8192 * 9), 256, 0, s>>>(pos, 3, x, 6, in0, (long)8192 * 9);
    mlp_bn(in0, FC0, h0, 8192, 16, 9, 0, 0.f, EPS_BN, stat, s);
    ws.off = mark;
  }
  // encoder blocks
  block(B1, h0, pos, 8192, 16, 16, x1f, ws, stat, s);  ws.off = mark;
  dec(x1f, 32, x1, 2048);  dec(pos, 3, pos1, 2048);
  {
    float* x2f = ws.f((size_t)2048 * 128);
    block(B2, x1, pos1, 2048, 32, 64, x2f, ws, stat, s);
    dec(x2f, 128, x2, 512);  dec(pos1, 3, pos2, 512);
    ws.off = mark;
  }
  {
    float* x3f = ws.f((size_t)512 * 256);
    block(B3, x2, pos2, 512, 128, 128, x3f, ws, stat, s);
    dec(x3f, 256, x3, 128);  dec(pos2, 3, pos3, 128);
    ws.off = mark;
  }
  {
    float* x4f = ws.f((size_t)128 * 512);
    block(B4, x3, pos3, 128, 256, 256, x4f, ws, stat, s);
    dec(x4f, 512, x4, 32);   dec(pos3, 3, pos4, 32);
    ws.off = mark;
  }
  // mid MLP (eps = 1e-5)
  mlp_bn(x4, MID, mm, 32, 512, 512, 1, 0.2f, 1e-5f, stat, s);
  // decoder (feature propagation, k=1 NN interp)
  fp_up(FP4, mm, 512, pos4, 32,   pos3, 128,  x3,  256, f4, 256, ws, stat, s); ws.off = mark;
  fp_up(FP3, f4, 256, pos3, 128,  pos2, 512,  x2,  128, f3, 128, ws, stat, s); ws.off = mark;
  fp_up(FP2, f3, 128, pos2, 512,  pos,  8192, x1f, 32,  f2, 32,  ws, stat, s); ws.off = mark;
  fp_up(FP1, f2, 32,  pos,  8192, pos,  8192, h0,  16,  f1, 8,   ws, stat, s); ws.off = mark;
  // head
  mlp_bn(f1, H1, g1, 8192, 64, 8, 1, 0.2f, EPS_BN, stat, s);
  mlp_bn(g1, H2, g2, 8192, 32, 64, 1, 0.2f, EPS_BN, stat, s);
  linear(g2, FCE.W, FCE.b, (float*)d_out, 8192, 7, 32, 0, 0.f, s);
}
